// CausalGraphTransformer_20761871908967
// MI455X (gfx1250) — matrix operand order repaired and checked by reading, not compiled, not run on hardware
//
#include <hip/hip_runtime.h>
#include <math.h>

#define NN   8192
#define EDG  98304
#define HID  256
#define FFD  1024
#define NLAY 4

typedef __attribute__((ext_vector_type(16))) __bf16 v16bf;
typedef __attribute__((ext_vector_type(8)))  float  v8f;
typedef __attribute__((ext_vector_type(4)))  float  f4;

__device__ __forceinline__ float gelu_f(float x) {
  return 0.5f * x * (1.0f + erff(x * 0.70710678118654752440f));
}

__device__ __forceinline__ void atomicMaxF(float* addr, float val) {
  int old = __float_as_int(*addr);
  while (__int_as_float(old) < val) {
    int prev = atomicCAS((int*)addr, old, __float_as_int(val));
    if (prev == old) break;
    old = prev;
  }
}

__device__ __forceinline__ v16bf load_afrag(const float* __restrict__ ap, int aKoff) {
  f4 x0 = *(const f4*)(ap + aKoff);
  f4 x1 = *(const f4*)(ap + aKoff + 4);
  f4 x2 = *(const f4*)(ap + 16 + aKoff);
  f4 x3 = *(const f4*)(ap + 16 + aKoff + 4);
  v16bf a;
#pragma unroll
  for (int i = 0; i < 4; ++i) {
    a[i]      = (__bf16)x0[i];
    a[4 + i]  = (__bf16)x1[i];
    a[8 + i]  = (__bf16)x2[i];
    a[12 + i] = (__bf16)x3[i];
  }
  return a;
}

// K-loop over [0,kcnt): 2 M-tiles x 2 N-tiles, B reused across M-tiles.
__device__ __forceinline__ void kloop(const float* __restrict__ ap0,
                                      const float* __restrict__ ap1,
                                      const __bf16* __restrict__ w0,
                                      const __bf16* __restrict__ w1,
                                      int kcnt, int aKoff,
                                      v8f& acc00, v8f& acc01,
                                      v8f& acc10, v8f& acc11) {
  for (int kb = 0; kb < kcnt; kb += 32) {
    v16bf a0 = load_afrag(ap0 + kb, aKoff);
    v16bf a1 = load_afrag(ap1 + kb, aKoff);
    typedef __attribute__((ext_vector_type(8))) __bf16 v8bf_;
    v8bf_ b0l = *(const v8bf_*)(w0 + kb), b0h = *(const v8bf_*)(w0 + kb + 16);
    v8bf_ b1l = *(const v8bf_*)(w1 + kb), b1h = *(const v8bf_*)(w1 + kb + 16);
    v16bf b0 = __builtin_shufflevector(b0l, b0h, 0,1,2,3,4,5,6,7,8,9,10,11,12,13,14,15);
    v16bf b1 = __builtin_shufflevector(b1l, b1h, 0,1,2,3,4,5,6,7,8,9,10,11,12,13,14,15);
    acc00 = __builtin_amdgcn_wmma_f32_16x16x32_bf16(false, a0, false, b0, (short)0, acc00, false, false);
    acc01 = __builtin_amdgcn_wmma_f32_16x16x32_bf16(false, a0, false, b1, (short)0, acc01, false, false);
    acc10 = __builtin_amdgcn_wmma_f32_16x16x32_bf16(false, a1, false, b0, (short)0, acc10, false, false);
    acc11 = __builtin_amdgcn_wmma_f32_16x16x32_bf16(false, a1, false, b1, (short)0, acc11, false, false);
  }
}

// ---------------------------------------------------------------------------
// Fused GEMM: C[M,Nc] = epilogue(A[M,K] @ W[K,Nc] + bias)
//  - A fp32 row-major (lda mult of 8, K mult of 32, zero-padded).
//  - Wt bf16 COLUMN-major (Wt[n*K + k]), zero-padded; Nc multiple of 256.
//  - Block = 256 threads (8 waves) computes 32 rows x 256 cols; each wave:
//    2 M-tiles x 2 N-tiles (4 WMMAs/chunk). grid.x = M/32, grid.y = Nc/256.
//  - Epilogue: +bias, +residual, LayerNorm (Nc==256), GELU.
//  - Gather mode (gsrc!=null): row m = concat(Xg[gsrc[m]], Xg[gdst[m]]), K=512.
// ---------------------------------------------------------------------------
__global__ __launch_bounds__(256)
void gemm_wmma(const float* __restrict__ A, int lda,
               const __bf16* __restrict__ Wt,
               const float* __restrict__ bias,
               float* __restrict__ C, int ldc, int coff,
               const float* __restrict__ res, int ldr,
               const float* __restrict__ lng, const float* __restrict__ lnb,
               int K, int do_gelu,
               const float* __restrict__ Xg,
               const int* __restrict__ gsrc, const int* __restrict__ gdst) {
  const int lane    = threadIdx.x & 31;
  const int wv      = threadIdx.x >> 5;
  const int r0      = blockIdx.x * 32;
  const int colbase = blockIdx.y * 256;
  const int m0      = r0 + (lane & 15);
  const int m1      = m0 + 16;
  const int aKoff   = (lane < 16) ? 0 : 8;   // A fragment K base (16-bit layout)
  const int bKoff   = (lane < 16) ? 0 : 8;   // B fragment K base (16-bit layout)

  const int n0 = colbase + wv * 32 + (lane & 15);
  const int n1 = n0 + 16;
  const __bf16* wp0 = Wt + (long)n0 * K + bKoff;
  const __bf16* wp1 = Wt + (long)n1 * K + bKoff;

  v8f acc00 = {}, acc01 = {}, acc10 = {}, acc11 = {};

  if (gsrc) {  // K == 512 gather mode: two clean half-loops, no per-iter select
    const float* pS0 = Xg + (long)gsrc[m0] * HID;
    const float* pS1 = Xg + (long)gsrc[m1] * HID;
    const float* pD0 = Xg + (long)gdst[m0] * HID;
    const float* pD1 = Xg + (long)gdst[m1] * HID;
    kloop(pS0, pS1, wp0,       wp1,       HID, aKoff, acc00, acc01, acc10, acc11);
    kloop(pD0, pD1, wp0 + HID, wp1 + HID, HID, aKoff, acc00, acc01, acc10, acc11);
  } else {
    const float* pA0 = A + (long)m0 * lda;
    const float* pA1 = A + (long)m1 * lda;
    kloop(pA0, pA1, wp0, wp1, K, aKoff, acc00, acc01, acc10, acc11);
  }

  const float b0v = bias ? bias[n0] : 0.0f;
  const float b1v = bias ? bias[n1] : 0.0f;

  if (lng == nullptr) {
    // direct epilogue: bias, residual, gelu
#pragma unroll
    for (int g = 0; g < 2; ++g) {
      const v8f aA = g ? acc10 : acc00;
      const v8f aB = g ? acc11 : acc01;
#pragma unroll
      for (int r = 0; r < 8; ++r) {
        int row = r0 + g * 16 + ((lane < 16) ? r : 8 + r);
        {
          float v = aA[r] + b0v;
          if (res)  v += res[(long)row * ldr + n0];
          if (do_gelu) v = gelu_f(v);
          C[(long)row * ldc + coff + n0] = v;
        }
        {
          float v = aB[r] + b1v;
          if (res)  v += res[(long)row * ldr + n1];
          if (do_gelu) v = gelu_f(v);
          C[(long)row * ldc + coff + n1] = v;
        }
      }
    }
  } else {
    // LayerNorm epilogue (Nc == 256, grid.y == 1; block owns full rows)
    __shared__ float sc[32][264];
    __shared__ float smean[32], srstd[32];
#pragma unroll
    for (int g = 0; g < 2; ++g) {
      const v8f aA = g ? acc10 : acc00;
      const v8f aB = g ? acc11 : acc01;
#pragma unroll
      for (int r = 0; r < 8; ++r) {
        int rl = g * 16 + ((lane < 16) ? r : 8 + r);
        {
          float v = aA[r] + b0v;
          if (res) v += res[(long)(r0 + rl) * ldr + n0];
          sc[rl][n0] = v;
        }
        {
          float v = aB[r] + b1v;
          if (res) v += res[(long)(r0 + rl) * ldr + n1];
          sc[rl][n1] = v;
        }
      }
    }
    __syncthreads();
    if (threadIdx.x < 32) {
      int rr = threadIdx.x;
      float s = 0.0f;
      for (int c = 0; c < 256; ++c) s += sc[rr][c];
      float mean = s * (1.0f / 256.0f);
      float v2 = 0.0f;
      for (int c = 0; c < 256; ++c) { float d = sc[rr][c] - mean; v2 += d * d; }
      smean[rr] = mean;
      srstd[rr] = rsqrtf(v2 * (1.0f / 256.0f) + 1e-5f);
    }
    __syncthreads();
    for (int idx = threadIdx.x; idx < 32 * 256; idx += 256) {
      int rr = idx >> 8, cc = idx & 255;
      float v = (sc[rr][cc] - smean[rr]) * srstd[rr] * lng[cc] + lnb[cc];
      if (do_gelu) v = gelu_f(v);
      C[(long)(r0 + rr) * ldc + coff + cc] = v;
    }
  }
}

// ---------------------------------------------------------------------------
// weight convert: Wt[n*Kpad + k] = (k<Ksrc && n<Nsrc) ? W[k*Nsrc + n] : 0
__global__ void k_wcvt(const float* __restrict__ W, __bf16* __restrict__ Wt,
                       int Ksrc, int Kpad, int Nsrc, int Npad) {
  int gid = blockIdx.x * 256 + threadIdx.x;
  if (gid >= Kpad * Npad) return;
  int n = gid / Kpad, k = gid % Kpad;
  float v = (k < Ksrc && n < Nsrc) ? W[(long)k * Nsrc + n] : 0.0f;
  Wt[gid] = (__bf16)v;
}

__global__ void k_fill(float* p, float v, int n) {
  int i = blockIdx.x * 256 + threadIdx.x;
  if (i < n) p[i] = v;
}

// zero-padded vector copy (for bo2)
__global__ void k_padvec(const float* __restrict__ s, float* __restrict__ d,
                         int nsrc, int npad) {
  int i = blockIdx.x * 256 + threadIdx.x;
  if (i < npad) d[i] = (i < nsrc) ? s[i] : 0.0f;
}

// pad edge attrs [E,8] -> [E,32] zero-padded
__global__ void k_epad(const float* __restrict__ ea, float* __restrict__ epad) {
  int gid = blockIdx.x * 256 + threadIdx.x;  // E*32
  int e = gid >> 5, c = gid & 31;
  epad[gid] = (c < 8) ? ea[e * 8 + c] : 0.0f;
}

// Build obs_in [N,320] (zero-padded concat) and masked realized [N,32]
__global__ void k_build(const float* __restrict__ obs, const float* __restrict__ realized,
                        const float* __restrict__ pkg,
                        const float* __restrict__ Eev, const float* __restrict__ Eloc,
                        const float* __restrict__ Epf, const float* __restrict__ Ereg,
                        const float* __restrict__ Ecar, const float* __restrict__ Eleg,
                        const float* __restrict__ Eship, const float* __restrict__ Epp,
                        const int* __restrict__ iev, const int* __restrict__ iloc,
                        const int* __restrict__ ipf, const int* __restrict__ ireg,
                        const int* __restrict__ icar, const int* __restrict__ ileg,
                        const int* __restrict__ iship,
                        const int* __restrict__ psrc, const int* __restrict__ pdst,
                        const int* __restrict__ positions, const int* __restrict__ cutoff,
                        float* __restrict__ obs_in, float* __restrict__ realm) {
  int n = blockIdx.x;
  float* row = obs_in + (long)n * 320;
  for (int c = threadIdx.x; c < 320; c += 128) {
    float v;
    if (c < 11) {
      v = obs[n * 11 + c];
    } else if (c < 235) {
      int cc = c - 11, g = cc >> 5, o = cc & 31;
      const float* tbl; int id;
      switch (g) {
        case 0: tbl = Eev;  id = iev[n];  break;
        case 1: tbl = Eloc; id = iloc[n]; break;
        case 2: tbl = Epf;  id = ipf[n];  break;
        case 3: tbl = Ereg; id = ireg[n]; break;
        case 4: tbl = Ecar; id = icar[n]; break;
        case 5: tbl = Eleg; id = ileg[n]; break;
        default: tbl = Eship; id = iship[n]; break;
      }
      v = tbl[id * 32 + o];
    } else if (c < 267) {
      v = Epp[psrc[n] * 32 + (c - 235)];
    } else if (c < 299) {
      v = Epp[pdst[n] * 32 + (c - 267)];
    } else if (c < 303) {
      v = pkg[n * 4 + (c - 299)];
    } else {
      v = 0.0f;
    }
    row[c] = v;
  }
  float mask = (positions[n] <= cutoff[0]) ? 1.0f : 0.0f;
  for (int c = threadIdx.x; c < 32; c += 128)
    realm[n * 32 + c] = (c < 20) ? realized[n * 20 + c] * mask : 0.0f;
}

// x += sinusoidal positional encoding
__global__ void k_addpe(float* __restrict__ x, const int* __restrict__ positions) {
  int gid = blockIdx.x * 256 + threadIdx.x;  // N*256 threads
  int n = gid >> 8, c = gid & 255;
  int p = positions[n];
  p = p < 0 ? 0 : (p > 199 ? 199 : p);
  int j = c >> 1;
  float ang = (float)p * __expf(-(float)(2 * j) * (9.210340371976184f / 256.0f));
  x[gid] += (c & 1) ? __cosf(ang) : __sinf(ang);
}

// per (edge, head): logits + segment-max into m
__global__ void k_logits(const float* __restrict__ q, const float* __restrict__ k,
                         const float* __restrict__ ee,
                         const int* __restrict__ src, const int* __restrict__ dst,
                         float* __restrict__ logits, float* __restrict__ mbuf) {
  int gid = blockIdx.x * 256 + threadIdx.x;  // E*8
  int e = gid >> 3, h = gid & 7;
  int s = src[e], d = dst[e];
  const float* qp = q  + (long)d * HID + h * 32;
  const float* kp = k  + (long)s * HID + h * 32;
  const float* ep = ee + (long)e * HID + h * 32;
  float acc = 0.0f;
#pragma unroll 8
  for (int i = 0; i < 32; ++i) acc += qp[i] * (kp[i] + ep[i]);
  acc *= 0.17677669529663687f;  // 1/sqrt(32)
  logits[gid] = acc;
  atomicMaxF(mbuf + d * 8 + h, acc);
}

// p = exp(logit - m[dst]); denom += p (segment-sum)
__global__ void k_expsum(const int* __restrict__ dst, float* __restrict__ logits,
                         const float* __restrict__ mbuf, float* __restrict__ denom) {
  int gid = blockIdx.x * 256 + threadIdx.x;  // E*8
  int e = gid >> 3, h = gid & 7;
  int d = dst[e];
  float p = __expf(logits[gid] - mbuf[d * 8 + h]);
  logits[gid] = p;
  atomicAdd(denom + d * 8 + h, p);
}

// out[dst] += (v[src] + ee) * alpha  (per channel)
__global__ void k_scatter(const float* __restrict__ v, const float* __restrict__ ee,
                          const float* __restrict__ pbuf, const float* __restrict__ denom,
                          const int* __restrict__ src, const int* __restrict__ dst,
                          float* __restrict__ out) {
  int gid = blockIdx.x * 256 + threadIdx.x;  // E*256
  int e = gid >> 8, c = gid & 255, h = c >> 5;
  int s = src[e], d = dst[e];
  float alpha = pbuf[e * 8 + h] / (denom[d * 8 + h] + 1e-16f);
  float msg = (v[(long)s * HID + c] + ee[(long)e * HID + c]) * alpha;
  atomicAdd(out + (long)d * HID + c, msg);
}

// beta gate + residual + LayerNorm1 : one block per node
__global__ __launch_bounds__(256)
void k_beta_ln(const float* __restrict__ x, const float* __restrict__ outb,
               const float* __restrict__ skip, const float* __restrict__ Wb,
               const float* __restrict__ g, const float* __restrict__ b,
               float* __restrict__ xout) {
  int n = blockIdx.x, c = threadIdx.x;
  __shared__ float red[256];
  float o  = outb[(long)n * HID + c];
  float xr = skip[(long)n * HID + c];
  float xv = x[(long)n * HID + c];
  red[c] = o * Wb[c] + xr * Wb[256 + c] + (o - xr) * Wb[512 + c];
  __syncthreads();
  for (int s = 128; s > 0; s >>= 1) { if (c < s) red[c] += red[c + s]; __syncthreads(); }
  float beta = 1.0f / (1.0f + __expf(-red[0]));
  __syncthreads();
  float y = xv + beta * xr + (1.0f - beta) * o;
  red[c] = y;
  __syncthreads();
  for (int s = 128; s > 0; s >>= 1) { if (c < s) red[c] += red[c + s]; __syncthreads(); }
  float mean = red[0] * (1.0f / 256.0f);
  __syncthreads();
  float dlt = y - mean;
  red[c] = dlt * dlt;
  __syncthreads();
  for (int s = 128; s > 0; s >>= 1) { if (c < s) red[c] += red[c + s]; __syncthreads(); }
  float rstd = rsqrtf(red[0] * (1.0f / 256.0f) + 1e-5f);
  xout[(long)n * HID + c] = dlt * rstd * g[c] + b[c];
}

// final projection: dout[e] = o2[e, 0:128] . Wo3 + bo3   (o2 row stride 256)
__global__ void k_final(const float* __restrict__ o2, const float* __restrict__ Wo3,
                        const float* __restrict__ bo3, float* __restrict__ dout) {
  int e = blockIdx.x * 256 + threadIdx.x;
  if (e >= EDG) return;
  const float* o = o2 + (long)e * 256;
  float s = bo3[0];
#pragma unroll 8
  for (int j = 0; j < 128; ++j) s += o[j] * Wo3[j];
  dout[e] = s;
}

// ---------------------------------------------------------------------------
extern "C" void kernel_launch(void* const* d_in, const int* in_sizes, int n_in,
                              void* d_out, int out_size, void* d_ws, size_t ws_size,
                              hipStream_t stream) {
  // float inputs
  const float* observable = (const float*)d_in[0];
  const float* realized   = (const float*)d_in[1];
  const float* pkg        = (const float*)d_in[2];
  const float* edge_attr  = (const float*)d_in[3];
  const float* Eev   = (const float*)d_in[4];
  const float* Eloc  = (const float*)d_in[5];
  const float* Epf   = (const float*)d_in[6];
  const float* Ereg  = (const float*)d_in[7];
  const float* Ecar  = (const float*)d_in[8];
  const float* Eleg  = (const float*)d_in[9];
  const float* Eship = (const float*)d_in[10];
  const float* Epp   = (const float*)d_in[11];
  const float* W_obs = (const float*)d_in[12]; const float* b_obs = (const float*)d_in[13];
  const float* lnog  = (const float*)d_in[14]; const float* lnob  = (const float*)d_in[15];
  const float* W_real= (const float*)d_in[16]; const float* b_real= (const float*)d_in[17];
  const float* lnrg  = (const float*)d_in[18]; const float* lnrb  = (const float*)d_in[19];
  const float* W_comb= (const float*)d_in[20]; const float* b_comb= (const float*)d_in[21];
  const float* lncg  = (const float*)d_in[22]; const float* lncb  = (const float*)d_in[23];
  const float* W_edge= (const float*)d_in[24]; const float* b_edge= (const float*)d_in[25];
  const float* lneg  = (const float*)d_in[26]; const float* lneb  = (const float*)d_in[27];
  const float* Wq    = (const float*)d_in[28];
  const float* Wk    = (const float*)d_in[29];
  const float* Wv    = (const float*)d_in[30];
  const float* We    = (const float*)d_in[31];
  const float* Wskip = (const float*)d_in[32];
  const float* bq    = (const float*)d_in[33];
  const float* bk    = (const float*)d_in[34];
  const float* bv    = (const float*)d_in[35];
  const float* be    = (const float*)d_in[36];
  const float* bskip = (const float*)d_in[37];
  const float* Wbeta = (const float*)d_in[38];
  const float* ln1g  = (const float*)d_in[39]; const float* ln1b = (const float*)d_in[40];
  const float* ln2g  = (const float*)d_in[41]; const float* ln2b = (const float*)d_in[42];
  const float* Wf1   = (const float*)d_in[43]; const float* bf1  = (const float*)d_in[44];
  const float* Wf2   = (const float*)d_in[45]; const float* bf2  = (const float*)d_in[46];
  const float* Wo1   = (const float*)d_in[47]; const float* bo1  = (const float*)d_in[48];
  const float* lnHg  = (const float*)d_in[49]; const float* lnHb = (const float*)d_in[50];
  const float* Wo2   = (const float*)d_in[51]; const float* bo2  = (const float*)d_in[52];
  const float* Wo3   = (const float*)d_in[53]; const float* bo3  = (const float*)d_in[54];
  // int inputs
  const int* ei        = (const int*)d_in[55];
  const int* src       = ei;
  const int* dst       = ei + EDG;
  const int* positions = (const int*)d_in[56];
  const int* iev   = (const int*)d_in[57];
  const int* iloc  = (const int*)d_in[58];
  const int* ipf   = (const int*)d_in[59];
  const int* ireg  = (const int*)d_in[60];
  const int* icar  = (const int*)d_in[61];
  const int* ileg  = (const int*)d_in[62];
  const int* iship = (const int*)d_in[63];
  const int* psrc  = (const int*)d_in[64];
  const int* pdst  = (const int*)d_in[65];
  const int* cutoff= (const int*)d_in[66];

  // ---- workspace carve-up ----
  float* ws = (float*)d_ws;
  size_t off = 0;
  auto alloc = [&](size_t n) { float* p = ws + off; off += n; return p; };
  float* OBS   = alloc((size_t)NN * 320);    // zero-padded K
  float* REALM = alloc((size_t)NN * 32);     // zero-padded K
  float* EPAD  = alloc((size_t)EDG * 32);    // zero-padded edge attrs
  float* HCAT  = alloc((size_t)NN * 512);
  float* XA    = alloc((size_t)NN * HID);
  float* XB    = alloc((size_t)NN * HID);
  float* QB    = alloc((size_t)NN * HID);
  float* KB    = alloc((size_t)NN * HID);
  float* VB    = alloc((size_t)NN * HID);
  float* SKB   = alloc((size_t)NN * HID);
  float* EBUF  = alloc((size_t)EDG * HID);   // e  (reused as o2 in head)
  float* EEBUF = alloc((size_t)EDG * HID);   // ee (reused as o1 in head)
  float* LOGB  = alloc((size_t)EDG * 8);
  float* MB    = alloc((size_t)NN * 8);
  float* DENB  = alloc((size_t)NN * 8);
  float* OUTB  = alloc((size_t)NN * HID);
  float* F1B   = alloc((size_t)NN * FFD);
  float* BO2P  = alloc((size_t)256);         // bo2 zero-padded to 256
  // bf16 weight area (column-major, zero-padded)
  __bf16* wb = (__bf16*)(ws + off);
  size_t boff = 0;
  auto balloc = [&](size_t n) { __bf16* p = wb + boff; boff += n; return p; };
  __bf16* WobsT  = balloc((size_t)256 * 320);
  __bf16* WrealT = balloc((size_t)256 * 32);
  __bf16* WcombT = balloc((size_t)256 * 512);
  __bf16* WedgeT = balloc((size_t)256 * 32);
  __bf16* WqT    = balloc((size_t)NLAY * 256 * 256);
  __bf16* WkT    = balloc((size_t)NLAY * 256 * 256);
  __bf16* WvT    = balloc((size_t)NLAY * 256 * 256);
  __bf16* WeT    = balloc((size_t)NLAY * 256 * 256);
  __bf16* WsT    = balloc((size_t)NLAY * 256 * 256);
  __bf16* Wf1T   = balloc((size_t)NLAY * 1024 * 256);
  __bf16* Wf2T   = balloc((size_t)NLAY * 256 * 1024);
  __bf16* Wo1T   = balloc((size_t)256 * 512);
  __bf16* Wo2T   = balloc((size_t)256 * 256);   // padded: 128 real + 128 zero cols

  auto wcvt = [&](const float* W, __bf16* Wt, int Ksrc, int Kpad, int Nsrc, int Npad) {
    int tot = Kpad * Npad;
    k_wcvt<<<(tot + 255) / 256, 256, 0, stream>>>(W, Wt, Ksrc, Kpad, Nsrc, Npad);
  };
  auto gemm = [&](const float* A, int lda, const __bf16* Wt, const float* bias,
                  float* C, int ldc, int coff, const float* res, int ldr,
                  const float* g, const float* b, int M, int K, int Nc, int do_gelu,
                  const float* Xg, const int* gs, const int* gd) {
    dim3 grid(M / 32, Nc / 256);
    gemm_wmma<<<grid, dim3(256), 0, stream>>>(A, lda, Wt, bias, C, ldc, coff,
                                              res, ldr, g, b, K, do_gelu,
                                              Xg, gs, gd);
  };

  // ---- stage -1: convert/transpose all weights to bf16 column-major ----
  wcvt(W_obs,  WobsT,  303, 320, 256, 256);
  wcvt(W_real, WrealT, 20,  32,  256, 256);
  wcvt(W_comb, WcombT, 512, 512, 256, 256);
  wcvt(W_edge, WedgeT, 8,   32,  256, 256);
  for (int l = 0; l < NLAY; ++l) {
    wcvt(Wq + (size_t)l * 65536, WqT + (size_t)l * 65536, 256, 256, 256, 256);
    wcvt(Wk + (size_t)l * 65536, WkT + (size_t)l * 65536, 256, 256, 256, 256);
    wcvt(Wv + (size_t)l * 65536, WvT + (size_t)l * 65536, 256, 256, 256, 256);
    wcvt(We + (size_t)l * 65536, WeT + (size_t)l * 65536, 256, 256, 256, 256);
    wcvt(Wskip + (size_t)l * 65536, WsT + (size_t)l * 65536, 256, 256, 256, 256);
    wcvt(Wf1 + (size_t)l * 262144, Wf1T + (size_t)l * 262144, 256,  256,  1024, 1024);
    wcvt(Wf2 + (size_t)l * 262144, Wf2T + (size_t)l * 262144, 1024, 1024, 256,  256);
  }
  wcvt(Wo1, Wo1T, 512, 512, 256, 256);
  wcvt(Wo2, Wo2T, 256, 256, 128, 256);       // pad output cols 128->256
  k_padvec<<<1, 256, 0, stream>>>(bo2, BO2P, 128, 256);

  // ---- stage 0: build padded inputs ----
  k_build<<<NN, 128, 0, stream>>>(observable, realized, pkg,
                                  Eev, Eloc, Epf, Ereg, Ecar, Eleg, Eship, Epp,
                                  iev, iloc, ipf, ireg, icar, ileg, iship,
                                  psrc, pdst, positions, cutoff, OBS, REALM);
  k_epad<<<EDG * 32 / 256, 256, 0, stream>>>(edge_attr, EPAD);

  // ---- stage 1: encoders ----
  gemm(OBS,   320, WobsT,  b_obs,  HCAT, 512, 0,   nullptr, 0, lnog, lnob, NN,  320, 256, 1, nullptr, nullptr, nullptr);
  gemm(REALM, 32,  WrealT, b_real, HCAT, 512, 256, nullptr, 0, lnrg, lnrb, NN,  32,  256, 1, nullptr, nullptr, nullptr);
  gemm(HCAT,  512, WcombT, b_comb, XA,   HID, 0,   nullptr, 0, lncg, lncb, NN,  512, 256, 1, nullptr, nullptr, nullptr);
  gemm(EPAD,  32,  WedgeT, b_edge, EBUF, HID, 0,   nullptr, 0, lneg, lneb, EDG, 32,  256, 1, nullptr, nullptr, nullptr);
  k_addpe<<<NN, 256, 0, stream>>>(XA, positions);

  // ---- stage 2: transformer layers ----
  for (int l = 0; l < NLAY; ++l) {
    gemm(XA, HID, WqT + (size_t)l * 65536, bq + l * HID,    QB,  HID, 0, nullptr, 0, nullptr, nullptr, NN, HID, 256, 0, nullptr, nullptr, nullptr);
    gemm(XA, HID, WkT + (size_t)l * 65536, bk + l * HID,    KB,  HID, 0, nullptr, 0, nullptr, nullptr, NN, HID, 256, 0, nullptr, nullptr, nullptr);
    gemm(XA, HID, WvT + (size_t)l * 65536, bv + l * HID,    VB,  HID, 0, nullptr, 0, nullptr, nullptr, NN, HID, 256, 0, nullptr, nullptr, nullptr);
    gemm(XA, HID, WsT + (size_t)l * 65536, bskip + l * HID, SKB, HID, 0, nullptr, 0, nullptr, nullptr, NN, HID, 256, 0, nullptr, nullptr, nullptr);
    gemm(EBUF, HID, WeT + (size_t)l * 65536, be + l * HID,  EEBUF, HID, 0, nullptr, 0, nullptr, nullptr, EDG, HID, 256, 0, nullptr, nullptr, nullptr);

    k_fill<<<(NN * 8 + 255) / 256, 256, 0, stream>>>(MB,   -1e30f, NN * 8);
    k_fill<<<(NN * 8 + 255) / 256, 256, 0, stream>>>(DENB, 0.0f,   NN * 8);
    k_fill<<<(NN * HID + 255) / 256, 256, 0, stream>>>(OUTB, 0.0f, NN * HID);

    k_logits<<<EDG * 8 / 256, 256, 0, stream>>>(QB, KB, EEBUF, src, dst, LOGB, MB);
    k_expsum<<<EDG * 8 / 256, 256, 0, stream>>>(dst, LOGB, MB, DENB);
    k_scatter<<<EDG, 256, 0, stream>>>(VB, EEBUF, LOGB, DENB, src, dst, OUTB);

    k_beta_ln<<<NN, 256, 0, stream>>>(XA, OUTB, SKB, Wbeta + l * 768,
                                      ln1g + l * HID, ln1b + l * HID, XB);

    gemm(XB,  HID, Wf1T + (size_t)l * 262144, bf1 + l * FFD, F1B, FFD, 0,
         nullptr, 0, nullptr, nullptr, NN, HID, 1024, 1, nullptr, nullptr, nullptr);
    gemm(F1B, FFD, Wf2T + (size_t)l * 262144, bf2 + l * HID, XA,  HID, 0,
         XB, HID, ln2g + l * HID, ln2b + l * HID, NN, FFD, 256, 0, nullptr, nullptr, nullptr);
  }

  // ---- stage 3: output head ----
  // o1 = gelu(ln(concat(x[src], x[dst]) @ Wo1 + bo1))  (gather mode, reuses EEBUF)
  gemm(nullptr, 0, Wo1T, bo1, EEBUF, HID, 0, nullptr, 0, lnHg, lnHb,
       EDG, 512, 256, 1, XA, src, dst);
  // o2 = gelu(o1 @ Wo2pad + bo2pad)  [E,256] (cols 128..255 are zero-weight junk)
  gemm(EEBUF, HID, Wo2T, BO2P, EBUF, 256, 0, nullptr, 0, nullptr, nullptr,
       EDG, HID, 256, 1, nullptr, nullptr, nullptr);
  // out = o2[:, :128] @ Wo3 + bo3  [E,1]
  k_final<<<(EDG + 255) / 256, 256, 0, stream>>>(EBUF, Wo3, bo3, (float*)d_out);

  (void)in_sizes; (void)n_in; (void)out_size; (void)ws_size;
}
